// MyModel_17179869184056
// MI455X (gfx1250) — compile-verified, matrix operand order checked by reading, not run on hardware
//
#include <hip/hip_runtime.h>
#include <hip/hip_bf16.h>

// ---------------------------------------------------------------------------
// Model constants
// ---------------------------------------------------------------------------
#define DMODEL 128
#define HEADS 4
#define DH 32
#define TFR 7
#define NIND 12
#define BATCH 256
#define JNT 17
#define LLAYERS 3
#define NSLOT 32
#define BN_EPS 1e-5f

typedef __bf16 bf16_t;
typedef __bf16 v16bf __attribute__((ext_vector_type(16)));
typedef __bf16 v8bf  __attribute__((ext_vector_type(8)));
typedef float  v8f   __attribute__((ext_vector_type(8)));

// ---------------------------------------------------------------------------
// fp32 -> bf16 weight conversion (weights tiny, done once per launch)
// ---------------------------------------------------------------------------
__global__ void f32_to_bf16_kernel(const float* __restrict__ src,
                                   bf16_t* __restrict__ dst, int nElem) {
    int stride = gridDim.x * blockDim.x;
    for (int i = blockIdx.x * blockDim.x + threadIdx.x; i < nElem; i += stride)
        dst[i] = (bf16_t)src[i];
}

// ---------------------------------------------------------------------------
// Joint embedding: h = leaky_relu(poses @ emb_w.T + emb_b)
// ---------------------------------------------------------------------------
__global__ void embed_kernel(const float* __restrict__ poses,
                             const float* __restrict__ ew,
                             const float* __restrict__ eb,
                             float* __restrict__ h, int Mrows) {
    long total = (long)Mrows * DMODEL;
    long stride = (long)gridDim.x * blockDim.x;
    for (long idx = (long)blockIdx.x * blockDim.x + threadIdx.x; idx < total; idx += stride) {
        int d = (int)(idx & (DMODEL - 1));
        long row = idx >> 7;
        float x = poses[row * 2 + 0];
        float y = poses[row * 2 + 1];
        float v = x * ew[d * 2 + 0] + y * ew[d * 2 + 1] + eb[d];
        h[idx] = v > 0.f ? v : 0.01f * v;
    }
}

// ---------------------------------------------------------------------------
// WMMA GEMM:  Y[M,N] = X[M,K] @ W^T (+bias) (+residual) (optional ReLU)
//   X fp32 row-major (staged once per block as a full 16xK bf16 strip in LDS),
//   W bf16 [N,K] row-major (== B^T, exactly the layout the B lanes want).
// One wave computes one 16x16 output tile; block covers 16 rows x N cols.
// blockDim.x = (N/16)*32, gridDim.x = M/16 (M is always a multiple of 16).
// K / RELU / RESID are compile-time so the k-loop fully unrolls (K/32
// back-to-back v_wmma, clause-able B loads, one barrier) and the epilogue is
// branch-free straight-line code.
//
// ISA lane layout (16-bit, K=32 step):
//   A: lane l holds row (l&15), K chunks { klo..klo+7 , klo+16..klo+23 },
//      klo = (l>>4)*8, packed ascending -> two 16B loads.
//   B: symmetric, lane l holds column (l&15) of the tile.
//   C/D: lane l, vgpr v -> row v + 8*(l>>4), col (l&15).
// ---------------------------------------------------------------------------
template <int K, bool RELU, bool RESID>
__global__ void __launch_bounds__(768) gemm_bf16_wmma(
        const float* __restrict__ X, const bf16_t* __restrict__ W,
        const float* __restrict__ bias, const float* __restrict__ residual,
        float* __restrict__ Y, int N)
{
    constexpr int PITCH = K + 8;   // bf16 elements; 16B-aligned rows, no bank conflicts
    __shared__ __attribute__((aligned(16))) bf16_t ldsA[16][PITCH];
    const int m0   = blockIdx.x * 16;
    const int tid  = threadIdx.x;
    const int wave = tid >> 5;
    const int lane = tid & 31;
    const int n0   = wave * 16;
    const int lrow = lane & 15;
    const int klo  = (lane >> 4) * 8;

    // stage the full 16 x K strip of X as bf16 (vectorized: float4 -> 4x bf16)
    constexpr int CHUNKS = 16 * (K / 4);
    for (int idx = tid; idx < CHUNKS; idx += blockDim.x) {
        int r = idx / (K / 4);
        int c = (idx % (K / 4)) * 4;
        const float4 v = *(const float4*)(X + (long)(m0 + r) * K + c);
        bf16_t* dst = &ldsA[r][c];
        dst[0] = (bf16_t)v.x;
        dst[1] = (bf16_t)v.y;
        dst[2] = (bf16_t)v.z;
        dst[3] = (bf16_t)v.w;
    }
    __syncthreads();

    v8f acc = {0.f, 0.f, 0.f, 0.f, 0.f, 0.f, 0.f, 0.f};
    const bf16_t* wrow = W + (long)(n0 + lrow) * K;
    __builtin_prefetch(wrow, 0, 3);

#pragma unroll
    for (int k0 = 0; k0 < K; k0 += 32) {
        v8bf alo = *(const v8bf*)&ldsA[lrow][k0 + klo];
        v8bf ahi = *(const v8bf*)&ldsA[lrow][k0 + klo + 16];
        v16bf a = __builtin_shufflevector(alo, ahi, 0,1,2,3,4,5,6,7,8,9,10,11,12,13,14,15);

        v8bf blo = *(const v8bf*)(wrow + k0 + klo);
        v8bf bhi = *(const v8bf*)(wrow + k0 + klo + 16);
        v16bf b = __builtin_shufflevector(blo, bhi, 0,1,2,3,4,5,6,7,8,9,10,11,12,13,14,15);

        acc = __builtin_amdgcn_wmma_f32_16x16x32_bf16(false, a, false, b,
                                                      (short)0, acc, false, false);
    }

    const int ncol  = n0 + lrow;
    const float bv  = bias[ncol];
    const int rbase = m0 + ((lane >> 4) * 8);

    float rv[8];
    if (RESID) {
        const float* rp = residual + (long)rbase * N + ncol;
#pragma unroll
        for (int v = 0; v < 8; ++v) rv[v] = rp[(long)v * N];
    }

    float* yp = Y + (long)rbase * N + ncol;
#pragma unroll
    for (int v = 0; v < 8; ++v) {
        float y = acc[v] + bv;
        if (RESID) y += rv[v];
        if (RELU)  y = fmaxf(y, 0.f);
        yp[(long)v * N] = y;
    }
}

// ---------------------------------------------------------------------------
// Per-graph dense MHSA (pre-output-projection). One block per graph, 128 thr:
// wave == head, lane == feature/key index. qkv: [M][384] (q|k|v).
// ---------------------------------------------------------------------------
__global__ void attn_kernel(const float* __restrict__ qkv,
                            float* __restrict__ out, int n) {
    __shared__ float sq[JNT][DMODEL], sk[JNT][DMODEL], sv[JNT][DMODEL];
    const int g   = blockIdx.x;
    const int tid = threadIdx.x;
    const long base = (long)g * n;

    for (int idx = tid; idx < n * DMODEL; idx += blockDim.x) {
        int r = idx >> 7, c = idx & (DMODEL - 1);
        long ro = (base + r) * (3 * DMODEL);
        sq[r][c] = qkv[ro + c];
        sk[r][c] = qkv[ro + DMODEL + c];
        sv[r][c] = qkv[ro + 2 * DMODEL + c];
    }
    __syncthreads();

    const int hd = tid >> 5;
    const int e  = tid & 31;
    const float scale = 0.17677669529663687f; // 1/sqrt(32)

    for (int i = 0; i < n; ++i) {
        float s = -1e30f;
        if (e < n) {
            float accd = 0.f;
            const float* qi = &sq[i][hd * DH];
            const float* ke = &sk[e][hd * DH];
#pragma unroll
            for (int t = 0; t < DH; ++t) accd += qi[t] * ke[t];
            s = accd * scale;
        }
        float m = s;
        for (int off = 16; off > 0; off >>= 1) m = fmaxf(m, __shfl_xor(m, off, 32));
        float ex = (e < n) ? __expf(s - m) : 0.f;
        float sum = ex;
        for (int off = 16; off > 0; off >>= 1) sum += __shfl_xor(sum, off, 32);
        float aL = ex / sum;

        float accv = 0.f;
        for (int j = 0; j < n; ++j) {
            float aj = __shfl(aL, j, 32);
            accv += aj * sv[j][hd * DH + e];
        }
        out[(base + i) * DMODEL + hd * DH + e] = accv;
    }
}

// ---------------------------------------------------------------------------
// t1 = graph-mean(xw) + h ; t2 = attn + h ; accumulate BN stats (slotted atomics)
// One block per graph, 128 threads (thread == channel).
// stats layout: [NSLOT][4][128] = sum1, sq1, sum2, sq2
// ---------------------------------------------------------------------------
__global__ void fuse_t1t2_kernel(const float* __restrict__ xw,
                                 const float* __restrict__ attn,
                                 const float* __restrict__ h,
                                 float* __restrict__ t1, float* __restrict__ t2,
                                 float* __restrict__ stats, int n) {
    const int g = blockIdx.x, d = threadIdx.x;
    const long base = (long)g * n * DMODEL + d;

    float ms = 0.f;
    for (int j = 0; j < n; ++j) ms += xw[base + (long)j * DMODEL];
    const float mean = ms / (float)n;

    float s1 = 0, q1 = 0, s2 = 0, q2 = 0;
    for (int j = 0; j < n; ++j) {
        long idx = base + (long)j * DMODEL;
        float hv = h[idx];
        float a = mean + hv;       t1[idx] = a; s1 += a; q1 += a * a;
        float b = attn[idx] + hv;  t2[idx] = b; s2 += b; q2 += b * b;
    }
    float* st = stats + (long)(g & (NSLOT - 1)) * (4 * DMODEL);
    atomicAdd(st + d, s1);
    atomicAdd(st + DMODEL + d, q1);
    atomicAdd(st + 2 * DMODEL + d, s2);
    atomicAdd(st + 3 * DMODEL + d, q2);
}

// ---------------------------------------------------------------------------
// BN stats over rows of x [M][128] (used for the MLP branch)
// ---------------------------------------------------------------------------
__global__ void stats_kernel(const float* __restrict__ x,
                             float* __restrict__ stats, int M) {
    const int d = threadIdx.x;
    float s = 0.f, q = 0.f;
    for (int row = blockIdx.x; row < M; row += gridDim.x) {
        float v = x[(long)row * DMODEL + d];
        s += v; q += v * v;
    }
    float* st = stats + (long)(blockIdx.x & (NSLOT - 1)) * (4 * DMODEL);
    atomicAdd(st + d, s);
    atomicAdd(st + DMODEL + d, q);
}

__global__ void zero_kernel(float* __restrict__ p, int nElem) {
    int stride = gridDim.x * blockDim.x;
    for (int i = blockIdx.x * blockDim.x + threadIdx.x; i < nElem; i += stride)
        p[i] = 0.f;
}

// finalize two BNs (t1 / t2): coef -> a1,b1,a2,b2 (each 128)
__global__ void finalize2_kernel(const float* __restrict__ stats,
                                 const float* __restrict__ gamma01,
                                 const float* __restrict__ beta01,
                                 float invM, float* __restrict__ coef) {
    const int d = threadIdx.x;
    float s1 = 0, q1 = 0, s2 = 0, q2 = 0;
    for (int sl = 0; sl < NSLOT; ++sl) {
        const float* st = stats + (long)sl * 4 * DMODEL;
        s1 += st[d];               q1 += st[DMODEL + d];
        s2 += st[2 * DMODEL + d];  q2 += st[3 * DMODEL + d];
    }
    float m1 = s1 * invM, v1 = q1 * invM - m1 * m1;
    float a1 = gamma01[d] * rsqrtf(v1 + BN_EPS);
    coef[d] = a1; coef[DMODEL + d] = beta01[d] - m1 * a1;
    float m2 = s2 * invM, v2 = q2 * invM - m2 * m2;
    float a2 = gamma01[DMODEL + d] * rsqrtf(v2 + BN_EPS);
    coef[2 * DMODEL + d] = a2; coef[3 * DMODEL + d] = beta01[DMODEL + d] - m2 * a2;
}

// finalize one BN (t3): coef -> a (128), b (128)
__global__ void finalize1_kernel(const float* __restrict__ stats,
                                 const float* __restrict__ gamma,
                                 const float* __restrict__ beta,
                                 float invM, float* __restrict__ coef) {
    const int d = threadIdx.x;
    float s = 0, q = 0;
    for (int sl = 0; sl < NSLOT; ++sl) {
        const float* st = stats + (long)sl * 4 * DMODEL;
        s += st[d]; q += st[DMODEL + d];
    }
    float m = s * invM, v = q * invM - m * m;
    float a = gamma[d] * rsqrtf(v + BN_EPS);
    coef[d] = a; coef[DMODEL + d] = beta[d] - m * a;
}

// out = BN1(t1) + BN2(t2)
__global__ void combine_kernel(const float* __restrict__ t1,
                               const float* __restrict__ t2,
                               const float* __restrict__ coef,
                               float* __restrict__ out, long total) {
    long stride = (long)gridDim.x * blockDim.x;
    for (long idx = (long)blockIdx.x * blockDim.x + threadIdx.x; idx < total; idx += stride) {
        int d = (int)(idx & (DMODEL - 1));
        out[idx] = coef[d] * t1[idx] + coef[DMODEL + d] +
                   coef[2 * DMODEL + d] * t2[idx] + coef[3 * DMODEL + d];
    }
}

// y = a[d]*x + b[d]
__global__ void affine_kernel(const float* __restrict__ x,
                              const float* __restrict__ coef,
                              float* __restrict__ y, long total) {
    long stride = (long)gridDim.x * blockDim.x;
    for (long idx = (long)blockIdx.x * blockDim.x + threadIdx.x; idx < total; idx += stride) {
        int d = (int)(idx & (DMODEL - 1));
        y[idx] = coef[d] * x[idx] + coef[DMODEL + d];
    }
}

// y[g] = mean over rowsPer consecutive rows of x
__global__ void pool_rows_kernel(const float* __restrict__ x,
                                 float* __restrict__ y, int rowsPer) {
    const int g = blockIdx.x, d = threadIdx.x;
    long base = (long)g * rowsPer * DMODEL + d;
    float s = 0.f;
    for (int j = 0; j < rowsPer; ++j) s += x[base + (long)j * DMODEL];
    y[(long)g * DMODEL + d] = s / (float)rowsPer;
}

// logits = pooled @ cls_w.T + cls_b  ([256,128] x [8,128])
__global__ void cls_kernel(const float* __restrict__ pooled,
                           const float* __restrict__ cw,
                           const float* __restrict__ cb,
                           float* __restrict__ out) {
    int idx = blockIdx.x * blockDim.x + threadIdx.x;
    if (idx >= BATCH * 8) return;
    int b = idx >> 3, c = idx & 7;
    float s = cb[c];
    for (int d = 0; d < DMODEL; ++d) s += pooled[b * DMODEL + d] * cw[c * DMODEL + d];
    out[idx] = s;
}

// ---------------------------------------------------------------------------
// Host orchestration
// ---------------------------------------------------------------------------
extern "C" void kernel_launch(void* const* d_in, const int* in_sizes, int n_in,
                              void* d_out, int out_size, void* d_ws, size_t ws_size,
                              hipStream_t stream) {
    const float* poses = (const float*)d_in[0];
    const float* emb_w = (const float*)d_in[1];
    const float* emb_b = (const float*)d_in[2];
    const float* cls_w = (const float*)d_in[3];
    const float* cls_b = (const float*)d_in[4];
    // per-stage params: gw gb qw qb ow ob bng bnb w1 b1 w2 b2
    const float* jp[12]; const float* ip[12];
    for (int i = 0; i < 12; ++i) { jp[i] = (const float*)d_in[5 + i]; ip[i] = (const float*)d_in[17 + i]; }

    const int G1 = BATCH * TFR * NIND;        // 21504 joint graphs
    const int M1 = G1 * JNT;                  // 365568 rows (multiple of 16)
    const int G2 = BATCH * TFR;               // 1792 individual graphs
    const int M2 = G2 * NIND;                 // 21504 rows (multiple of 16)

    // ---- workspace layout ----
    char* ws = (char*)d_ws;
    const size_t sz128 = (size_t)M1 * DMODEL * sizeof(float);
    float* h1   = (float*)(ws + 0 * sz128);
    float* t1   = (float*)(ws + 1 * sz128);
    float* t2   = (float*)(ws + 2 * sz128);
    float* bufA = (float*)(ws + 3 * sz128);
    float* bufB = (float*)(ws + 4 * sz128);
    char*  qkvR = ws + 5 * sz128;                        // M1 x 384 fp32 region
    const size_t szqkv = (size_t)M1 * 3 * DMODEL * sizeof(float);
    float* qkv1 = (float*)qkvR;
    // stage-2 buffers alias the (then dead) qkv1 region
    const size_t sz128b = (size_t)M2 * DMODEL * sizeof(float);
    float* h2    = (float*)(qkvR + 0 * sz128b);
    float* t1b   = (float*)(qkvR + 1 * sz128b);
    float* t2b   = (float*)(qkvR + 2 * sz128b);
    float* bufA2 = (float*)(qkvR + 3 * sz128b);
    float* bufB2 = (float*)(qkvR + 4 * sz128b);
    float* qkv2  = (float*)(qkvR + 5 * sz128b);
    // tail: stats / coef / pooled / bf16 weights
    char* tail = qkvR + szqkv;
    float* stats  = (float*)tail;                 tail += (size_t)NSLOT * 4 * DMODEL * sizeof(float);
    float* coef   = (float*)tail;                 tail += 6 * DMODEL * sizeof(float);
    float* pooled = (float*)tail;                 tail += (size_t)BATCH * DMODEL * sizeof(float);
    bf16_t* wbf   = (bf16_t*)tail;

    // bf16 weight slots (elements): gw, qw, ow, w1, w2 per stage
    const int nGW = LLAYERS * DMODEL * DMODEL;        // 49152
    const int nQW = LLAYERS * 3 * DMODEL * DMODEL;    // 147456
    const int nW1 = LLAYERS * 2 * DMODEL * DMODEL;    // 98304
    bf16_t* wb[2][5];
    bf16_t* p = wbf;
    for (int s = 0; s < 2; ++s) {
        wb[s][0] = p; p += nGW;   // gw
        wb[s][1] = p; p += nQW;   // qw
        wb[s][2] = p; p += nGW;   // ow
        wb[s][3] = p; p += nW1;   // w1
        wb[s][4] = p; p += nW1;   // w2
    }
    {
        const float* src[2][5] = { { jp[0], jp[2], jp[4], jp[8], jp[10] },
                                   { ip[0], ip[2], ip[4], ip[8], ip[10] } };
        const int cnt[5] = { nGW, nQW, nGW, nW1, nW1 };
        for (int s = 0; s < 2; ++s)
            for (int t = 0; t < 5; ++t)
                f32_to_bf16_kernel<<<(cnt[t] + 255) / 256, 256, 0, stream>>>(src[s][t], wb[s][t], cnt[t]);
    }

    // GEMM dispatch: plain (K=128), relu (K=128), residual (K=256)
    auto gemm128 = [&](const float* X, const bf16_t* W, const float* bias,
                       float* Y, int M, int N) {
        gemm_bf16_wmma<128, false, false><<<dim3(M / 16), dim3((N / 16) * 32), 0, stream>>>(
            X, W, bias, nullptr, Y, N);
    };
    auto gemm128_relu = [&](const float* X, const bf16_t* W, const float* bias,
                            float* Y, int M, int N) {
        gemm_bf16_wmma<128, true, false><<<dim3(M / 16), dim3((N / 16) * 32), 0, stream>>>(
            X, W, bias, nullptr, Y, N);
    };
    auto gemm256_resid = [&](const float* X, const bf16_t* W, const float* bias,
                             const float* resid, float* Y, int M, int N) {
        gemm_bf16_wmma<256, false, true><<<dim3(M / 16), dim3((N / 16) * 32), 0, stream>>>(
            X, W, bias, resid, Y, N);
    };

    const int statsN = NSLOT * 4 * DMODEL;

    auto run_stage = [&](float* h, float* T1, float* T2, float* A, float* B, float* QKV,
                         bf16_t* const* W, const float* const* prm, int M, int n, int G) {
        const float* gb  = prm[1];
        const float* qb  = prm[3];
        const float* ob  = prm[5];
        const float* bng = prm[6];
        const float* bnb = prm[7];
        const float* b1  = prm[9];
        const float* b2  = prm[11];
        const float invM = 1.f / (float)M;
        const long total = (long)M * DMODEL;
        const int  ethr  = 256;
        const int  egr   = (int)((total + ethr - 1) / ethr);

        for (int l = 0; l < LLAYERS; ++l) {
            // GCN branch: xw = h @ gw^T + gb -> A
            gemm128(h, W[0] + (size_t)l * DMODEL * DMODEL, gb + l * DMODEL, A, M, DMODEL);
            // qkv = h @ qw^T + qb
            gemm128(h, W[1] + (size_t)l * 3 * DMODEL * DMODEL, qb + l * 3 * DMODEL, QKV, M, 3 * DMODEL);
            // per-graph MHSA -> B
            attn_kernel<<<G, 128, 0, stream>>>(QKV, B, n);
            // output projection (in place on B; each block owns its 16-row slice)
            gemm128(B, W[2] + (size_t)l * DMODEL * DMODEL, ob + l * DMODEL, B, M, DMODEL);
            // t1/t2 + stats
            zero_kernel<<<(statsN + 255) / 256, 256, 0, stream>>>(stats, statsN);
            fuse_t1t2_kernel<<<G, 128, 0, stream>>>(A, B, h, T1, T2, stats, n);
            finalize2_kernel<<<1, 128, 0, stream>>>(stats, bng + (size_t)l * 3 * DMODEL,
                                                    bnb + (size_t)l * 3 * DMODEL, invM, coef);
            // out = BN(t1) + BN(t2) -> A
            combine_kernel<<<egr, ethr, 0, stream>>>(T1, T2, coef, A, total);
            // MLP: m1 = relu(out @ w1^T + b1) -> QKV (M x 256 fits in M x 384)
            gemm128_relu(A, W[3] + (size_t)l * 2 * DMODEL * DMODEL, b1 + l * 2 * DMODEL, QKV, M, 2 * DMODEL);
            // t3 = out + m1 @ w2^T + b2 -> B
            gemm256_resid(QKV, W[4] + (size_t)l * 2 * DMODEL * DMODEL, b2 + l * DMODEL, A, B, M, DMODEL);
            // BN(t3) -> h (next layer input)
            zero_kernel<<<(statsN + 255) / 256, 256, 0, stream>>>(stats, statsN);
            stats_kernel<<<2048, 128, 0, stream>>>(B, stats, M);
            finalize1_kernel<<<1, 128, 0, stream>>>(stats, bng + (size_t)l * 3 * DMODEL + 2 * DMODEL,
                                                    bnb + (size_t)l * 3 * DMODEL + 2 * DMODEL,
                                                    invM, coef + 4 * DMODEL);
            affine_kernel<<<egr, ethr, 0, stream>>>(B, coef + 4 * DMODEL, h, total);
        }
    };

    // ---- pipeline ----
    {
        long total1 = (long)M1 * DMODEL;
        embed_kernel<<<(int)((total1 + 255) / 256), 256, 0, stream>>>(poses, emb_w, emb_b, h1, M1);
    }
    run_stage(h1, t1, t2, bufA, bufB, qkv1, wb[0], jp, M1, JNT, G1);
    // mean over joints: [G1 graphs of 17 rows] -> h2 [M2 rows]
    pool_rows_kernel<<<G1, 128, 0, stream>>>(h1, h2, JNT);
    run_stage(h2, t1b, t2b, bufA2, bufB2, qkv2, wb[1], ip, M2, NIND, G2);
    // mean over (T, N): 84 rows per batch element
    pool_rows_kernel<<<BATCH, 128, 0, stream>>>(h2, pooled, TFR * NIND);
    cls_kernel<<<8, 256, 0, stream>>>(pooled, cls_w, cls_b, (float*)d_out);
}
